// DeepAttensionModule_26388279066953
// MI455X (gfx1250) — matrix operand order repaired and checked by reading, not compiled, not run on hardware
//
#include <hip/hip_runtime.h>
#include <hip/hip_bf16.h>

// ---------------------------------------------------------------------------
// DeepAttention module on gfx1250 (MI455X). wave32, WMMA f32_16x16x32_f16.
// All GEMM operands are stored pre-swizzled in WMMA fragment layout so every
// fragment load is two 128-bit loads per lane (no scalar gather/packing).
//   prep  : fp32 weights -> swizzled fp16 B-fragments; fold W_k @ W_q / 4
//   encode: time-enc + one-hot -> combined (f16 A-fragments, K pad 31->32)
//   mlp2  : psi & phi MLPs via WMMA (16-token tiles, LDS fragment relay)
//   scanA : chunked cumsum/count scan (8-way decoupled per batch)
//   keyqk : agg @ arho (WMMA) + preattn = [combined,agg'] @ Wfold (WMMA)
//   scanB : chunked online-softmax scan per (b,h) -> agg2 A-fragments
//   rho   : 128->64->64 MLP via WMMA -> output f32
// ---------------------------------------------------------------------------

typedef __attribute__((ext_vector_type(16))) _Float16 v16h;
typedef __attribute__((ext_vector_type(8)))  float    v8f;

#define BB 4
#define PP 2048
#define TOK (BB * PP)
#define HEADS 4
#define DOT 16
#define PHI 32
#define AGG2 128
#define NTILES (TOK / 16)   /* 512 */

// Fragment table offsets (halves). One fragment = 32 lanes x 16 halves = 512.
#define F_PSI1  0
#define F_PSI2  (F_PSI1 + 2 * 512)
#define F_PHI1  (F_PSI2 + 2 * 512)
#define F_PHI2  (F_PHI1 + 2 * 512)
#define F_ARHO  (F_PHI2 + 2 * 512)
#define F_WFOLD (F_ARHO + 2 * 512)
#define F_RHO1  (F_WFOLD + 2 * 512)
#define F_RHO2  (F_RHO1 + 16 * 512)
#define F_TOTAL (F_RHO2 + 8 * 512)

#define ZERO8F {0.f, 0.f, 0.f, 0.f, 0.f, 0.f, 0.f, 0.f}

// Load one pre-swizzled fragment: two 128-bit loads per lane.
__device__ inline v16h load_frag(const _Float16* fragBase) {
  return *(const v16h*)(fragBase + (threadIdx.x & 31) * 16);
}

// Position of element (m, k) inside a 16x32 f16 A-fragment (ISA 7.12.2):
// lane = m + 16*k[3]; elem = k[2:0] + 8*k[4].
__device__ inline int afrag_pos(int m, int k) {
  int lane = (m & 15) + (((k >> 3) & 1) << 4);
  int e = (k & 7) + ((k >> 4) << 3);
  return lane * 16 + e;
}

__device__ inline v8f wmma16(v16h a, v16h b, v8f c) {
  return __builtin_amdgcn_wmma_f32_16x16x32_f16(
      /*neg_a=*/false, a, /*neg_b=*/false, b,
      /*c_mod=*/(short)0, c, /*reuse_a=*/false, /*reuse_b=*/false);
}

// ---- prep: swizzle weights into B-fragment layout --------------------------
// B fragment (32x16): lane holds col n = n0 + lane[3:0], rows k0 + 8*lane[4]
// + elem[2:0] + 16*elem[3]. Fragments ordered f = nt*(K/32) + kt.
__device__ inline void swz_weight(const float* W, int ldn, int K, int Kreal,
                                  int N, _Float16* dst, int t) {
  int ktn = K >> 5;
  int total = (N >> 4) * ktn * 512;
  for (int idx = t; idx < total; idx += 256) {
    int f = idx >> 9;
    int r = idx & 511;
    int lane = r >> 4, e = r & 15;
    int nt = f / ktn, kt = f - nt * ktn;
    int n = nt * 16 + (lane & 15);
    int k = kt * 32 + ((lane >> 4) << 3) + (e & 7) + ((e >> 3) << 4);
    float v = (k < Kreal) ? W[k * ldn + n] : 0.f;
    dst[idx] = (_Float16)v;
  }
}

__global__ void __launch_bounds__(256) prep_weights(
    const float* psi_w1, const float* psi_w2,
    const float* phi_w1, const float* phi_w2,
    const float* arho_w, const float* W_k, const float* W_q,
    const float* rho_w1, const float* rho_w2, _Float16* wf) {
  __shared__ float wfold[64 * 16];
  int t = threadIdx.x;
  // Wfold (64x16): rows 0..30 combined dims, row 31 pad, rows 32..63 agg dims.
  for (int i = t; i < 64 * 16; i += 256) {
    int r = i >> 4, h = i & 15;
    float v = 0.f;
    if (h < HEADS && r != 31) {
      int src = (r < 31) ? r : r - 1;
#pragma unroll
      for (int d = 0; d < DOT; ++d)
        v += W_k[src * (DOT * HEADS) + d * HEADS + h] * W_q[h * DOT + d];
      v *= 0.25f; // 1/sqrt(DOT)
    }
    wfold[i] = v;
  }
  __syncthreads();
  swz_weight(psi_w1, 32, 32, 31, 32, wf + F_PSI1, t);
  swz_weight(psi_w2, 32, 32, 32, 32, wf + F_PSI2, t);
  swz_weight(phi_w1, 32, 32, 31, 32, wf + F_PHI1, t);
  swz_weight(phi_w2, 32, 32, 32, 32, wf + F_PHI2, t);
  swz_weight(arho_w, 32, 32, 32, 32, wf + F_ARHO, t);
  swz_weight(wfold, 16, 64, 64, 16, wf + F_WFOLD, t);
  swz_weight(rho_w1, 64, 128, 128, 64, wf + F_RHO1, t);
  swz_weight(rho_w2, 64, 64, 64, 64, wf + F_RHO2, t);
}

// ---- encode: combined features -> A-fragment tiles (f16) -------------------
__global__ void __launch_bounds__(256) encode_kernel(
    const float* times, const float* values, const int* meas,
    const float* mask, _Float16* comb_swz) {
  int t = blockIdx.x * blockDim.x + threadIdx.x;
  if (t >= TOK) return;
  float tm = times[t], val = values[t], mk = mask[t];
  int mm = meas[t];
  // pos_vec = 10000^{0,0,.25,.25,.5,.5,.75,.75} = {1,1,10,10,100,100,1000,1000}
  const float inv[8] = {1.f, 1.f, 0.1f, 0.1f, 0.01f, 0.01f, 0.001f, 0.001f};
  _Float16 o[32];
#pragma unroll
  for (int i = 0; i < 8; ++i) {
    float r = tm * inv[i];
    float v = (i & 1) ? __cosf(r) : __sinf(r);
    o[i] = (_Float16)(v * mk);
  }
  o[8] = (_Float16)(val * mk);
#pragma unroll
  for (int j = 0; j < 22; ++j)
    o[9 + j] = (_Float16)(((mm == j + 1) ? 1.f : 0.f) * mk);
  o[31] = (_Float16)0.f;

  // scatter into A-fragment layout, 2 halves (one dword) at a time
  int tile = t >> 4, m = t & 15;
  unsigned* dst = (unsigned*)comb_swz + tile * 256;
#pragma unroll
  for (int k2 = 0; k2 < 16; ++k2) {
    int lane = m + (((k2 >> 2) & 1) << 4);
    int e2 = (k2 & 3) + ((k2 >> 3) << 2);
    unsigned pv;
    __builtin_memcpy(&pv, &o[2 * k2], 4);
    dst[lane * 8 + e2] = pv;
  }
}

// ---- psi & phi MLPs (WMMA), 4 waves/block, one 16-token tile per wave ------
__global__ void __launch_bounds__(128) mlp_psi_phi(
    const _Float16* comb_swz, const float* mask, const _Float16* wf,
    const float* psi_b1, const float* psi_b2,
    const float* phi_b1, const float* phi_b2,
    float* enc_psi, float* enc_phi) {
  __shared__ __attribute__((aligned(32))) _Float16 scr[4][512];
  int wave = threadIdx.x >> 5, lane = threadIdx.x & 31;
  int tile = blockIdx.x * 4 + wave;
  int tb = tile * 16;
  int n = lane & 15;
  int mBase = (lane < 16) ? 0 : 8;

  v16h a0 = load_frag(comb_swz + tile * 512);

#pragma unroll
  for (int which = 0; which < 2; ++which) {
    int W1 = which ? F_PHI1 : F_PSI1;
    int W2 = which ? F_PHI2 : F_PSI2;
    const float* b1 = which ? phi_b1 : psi_b1;
    const float* b2 = which ? phi_b2 : psi_b2;
    float* outp = which ? enc_phi : enc_psi;

    // layer 1 -> relu -> LDS fragment relay
#pragma unroll
    for (int nt = 0; nt < 2; ++nt) {
      v16h bw = load_frag(wf + W1 + nt * 512);
      v8f c = ZERO8F;
      c = wmma16(a0, bw, c);
      float bias = b1[nt * 16 + n];
#pragma unroll
      for (int i = 0; i < 8; ++i) {
        float h = c[i] + bias;
        h = h > 0.f ? h : 0.f;
        scr[wave][afrag_pos(mBase + i, nt * 16 + n)] = (_Float16)h;
      }
    }
    __syncthreads();
    v16h a1 = load_frag(scr[wave]);
    __syncthreads();

    // layer 2 -> relu * mask -> global f32 (row-major; scans read row-major)
#pragma unroll
    for (int nt = 0; nt < 2; ++nt) {
      v16h bw = load_frag(wf + W2 + nt * 512);
      v8f c = ZERO8F;
      c = wmma16(a1, bw, c);
      float bias = b2[nt * 16 + n];
#pragma unroll
      for (int i = 0; i < 8; ++i) {
        int m = tb + mBase + i;
        float h = c[i] + bias;
        h = h > 0.f ? h : 0.f;
        outp[m * 32 + nt * 16 + n] = h * mask[m];
      }
    }
    __syncthreads();
  }
}

// ---- chunked cumsum/count scan: 1 block per batch, 8 warp-chunks -----------
__global__ void __launch_bounds__(256) scan_agg(
    const float* enc_psi, const float* mask, _Float16* agg_swz) {
  const int CH = PP / 8;
  __shared__ float csum[8][32];
  int b = blockIdx.x;
  int lane = threadIdx.x & 31, w = threadIdx.x >> 5;
  int base = b * PP + w * CH;
  // per-lane feature index is fixed: A-fragment coords are constants
  int laneOff = ((lane >> 3) & 1) << 4;
  int e = (lane & 7) + ((lane >> 4) << 3);

  float acc = 0.f;
  for (int p = 0; p < CH; ++p) acc += enc_psi[(base + p) * 32 + lane];
  csum[w][lane] = acc;
  __syncthreads();

  float pre = 0.f;
  for (int j = 0; j < w; ++j) pre += csum[j][lane];

  acc = pre;
  for (int p = 0; p < CH; ++p) {
    int t = base + p;
    acc += enc_psi[t * 32 + lane];
    float agg = acc / (float)(w * CH + p + 1) * mask[t];
    agg_swz[(t >> 4) * 512 + ((t & 15) + laneOff) * 16 + e] = (_Float16)agg;
    if (p + 4 < CH) __builtin_prefetch(&enc_psi[(t + 4) * 32 + lane], 0, 0);
  }
}

// ---- agg@arho (WMMA) + preattn = [combined, agg'] @ Wfold (WMMA) -----------
__global__ void __launch_bounds__(128) key_preattn(
    const _Float16* comb_swz, const _Float16* agg_swz, const float* mask,
    const _Float16* wf, const float* arho_b, float* preattn) {
  __shared__ __attribute__((aligned(32))) _Float16 scr[4][512];
  int wave = threadIdx.x >> 5, lane = threadIdx.x & 31;
  int tile = blockIdx.x * 4 + wave;
  int tb = tile * 16;
  int n = lane & 15;
  int mBase = (lane < 16) ? 0 : 8;

  // comb2 K=0..31 fragment IS the swizzled combined tile (no staging copy)
  v16h aComb = load_frag(comb_swz + tile * 512);

  // agg projection (no relu): (16x32)@(32x32) + bias, * mask -> LDS fragment
  v16h aA = load_frag(agg_swz + tile * 512);
#pragma unroll
  for (int nt = 0; nt < 2; ++nt) {
    v16h bw = load_frag(wf + F_ARHO + nt * 512);
    v8f c = ZERO8F;
    c = wmma16(aA, bw, c);
    float bias = arho_b[nt * 16 + n];
#pragma unroll
    for (int i = 0; i < 8; ++i) {
      int m = mBase + i;
      float v = (c[i] + bias) * mask[tb + m];
      scr[wave][afrag_pos(m, nt * 16 + n)] = (_Float16)v;
    }
  }
  __syncthreads();
  v16h aProj = load_frag(scr[wave]);

  // preattn = comb2(16x64) @ Wfold(64x16), only cols 0..3 valid
  v16h bf0 = load_frag(wf + F_WFOLD);
  v16h bf1 = load_frag(wf + F_WFOLD + 512);
  v8f c = ZERO8F;
  c = wmma16(aComb, bf0, c);
  c = wmma16(aProj, bf1, c);
  if (n < HEADS) {
#pragma unroll
    for (int i = 0; i < 8; ++i) {
      int m = tb + mBase + i;
      preattn[m * HEADS + n] = c[i] * mask[m];
    }
  }
}

// ---- chunked online-softmax scan per (b,h): (m, den, num) is associative ---
__global__ void __launch_bounds__(256) scan_attn(
    const float* preattn, const float* enc_phi, const float* mask,
    _Float16* agg2_swz) {
  const int CH = PP / 8;
  __shared__ float cm[8], cden[8], cnum[8][32];
  int b = blockIdx.x >> 2, h = blockIdx.x & 3;
  int lane = threadIdx.x & 31, w = threadIdx.x >> 5;
  int base = b * PP + w * CH;
  const float NEG_INF = -__builtin_inff();
  // agg2 feature f = h*32 + lane -> fragment h, k = lane (constants per lane)
  int laneOff = ((lane >> 3) & 1) << 4;
  int e = (lane & 7) + ((lane >> 4) << 3);

  // pass 1: chunk-local state from identity
  float M = NEG_INF, D = 0.f, N = 0.f;
  for (int p = 0; p < CH; ++p) {
    int t = base + p;
    float s = preattn[t * HEADS + h];
    float mn = s > M ? s : M;
    float sc = __expf(M - mn);
    float pe = __expf(s - mn);
    D = D * sc + pe;
    N = N * sc + pe * enc_phi[t * 32 + lane];
    M = mn;
  }
  if (lane == 0) { cm[w] = M; cden[w] = D; }
  cnum[w][lane] = N;
  __syncthreads();

  // pass 2: exclusive combine of chunks 0..w-1
  M = NEG_INF; D = 0.f; N = 0.f;
  for (int j = 0; j < w; ++j) {
    float mj = cm[j];
    float mn = mj > M ? mj : M;
    float s0 = __expf(M - mn), s1 = __expf(mj - mn);
    D = D * s0 + cden[j] * s1;
    N = N * s0 + cnum[j][lane] * s1;
    M = mn;
  }

  // pass 3: re-scan chunk emitting per-position outputs into A-fragments
  for (int p = 0; p < CH; ++p) {
    int t = base + p;
    float s = preattn[t * HEADS + h];
    float mn = s > M ? s : M;
    float sc = __expf(M - mn);
    float pe = __expf(s - mn);
    D = D * sc + pe;
    N = N * sc + pe * enc_phi[t * 32 + lane];
    M = mn;
    float o = N / D * mask[t];
    agg2_swz[(t >> 4) * 2048 + h * 512 + ((t & 15) + laneOff) * 16 + e] =
        (_Float16)o;
    if (p + 4 < CH) __builtin_prefetch(&enc_phi[(t + 4) * 32 + lane], 0, 0);
  }
}

// ---- rho MLP: (16x128)@(128x64) relu, (16x64)@(64x64) relu * mask ----------
__global__ void __launch_bounds__(128) rho_mlp(
    const _Float16* agg2_swz, const float* mask, const _Float16* wf,
    const float* rho_b1, const float* rho_b2, float* out) {
  __shared__ __attribute__((aligned(32))) _Float16 scr[4][1024];
  int wave = threadIdx.x >> 5, lane = threadIdx.x & 31;
  int tile = blockIdx.x * 4 + wave;
  int tb = tile * 16;
  int n = lane & 15;
  int mBase = (lane < 16) ? 0 : 8;

  v16h a[4];
#pragma unroll
  for (int kk = 0; kk < 4; ++kk)
    a[kk] = load_frag(agg2_swz + tile * 2048 + kk * 512);

#pragma unroll
  for (int nt = 0; nt < 4; ++nt) {
    v8f c = ZERO8F;
#pragma unroll
    for (int kk = 0; kk < 4; ++kk) {
      v16h bw = load_frag(wf + F_RHO1 + (nt * 4 + kk) * 512);
      c = wmma16(a[kk], bw, c);
    }
    float bias = rho_b1[nt * 16 + n];
    int k = nt * 16 + n;               // output feature -> relay coords
    int fr = (k >> 5) * 512;
    int kk31 = k & 31;
#pragma unroll
    for (int i = 0; i < 8; ++i) {
      float h = c[i] + bias;
      h = h > 0.f ? h : 0.f;
      scr[wave][fr + afrag_pos(mBase + i, kk31)] = (_Float16)h;
    }
  }
  __syncthreads();
  v16h a2[2];
  a2[0] = load_frag(scr[wave]);
  a2[1] = load_frag(scr[wave] + 512);
  __syncthreads();

#pragma unroll
  for (int nt = 0; nt < 4; ++nt) {
    v8f c = ZERO8F;
#pragma unroll
    for (int kk = 0; kk < 2; ++kk) {
      v16h bw = load_frag(wf + F_RHO2 + (nt * 2 + kk) * 512);
      c = wmma16(a2[kk], bw, c);
    }
    float bias = rho_b2[nt * 16 + n];
#pragma unroll
    for (int i = 0; i < 8; ++i) {
      int m = tb + mBase + i;
      float h = c[i] + bias;
      h = h > 0.f ? h : 0.f;
      out[m * 64 + nt * 16 + n] = h * mask[m];
    }
  }
}

// ---------------------------------------------------------------------------
extern "C" void kernel_launch(void* const* d_in, const int* in_sizes, int n_in,
                              void* d_out, int out_size, void* d_ws,
                              size_t ws_size, hipStream_t stream) {
  (void)in_sizes; (void)n_in; (void)out_size; (void)ws_size;
  const float* times   = (const float*)d_in[0];
  const float* values  = (const float*)d_in[1];
  const int*   meas    = (const int*)d_in[2];
  const float* mask    = (const float*)d_in[3];
  const float* psi_w1  = (const float*)d_in[4];
  const float* psi_b1  = (const float*)d_in[5];
  const float* psi_w2  = (const float*)d_in[6];
  const float* psi_b2  = (const float*)d_in[7];
  const float* arho_w  = (const float*)d_in[8];
  const float* arho_b  = (const float*)d_in[9];
  const float* W_k     = (const float*)d_in[10];
  const float* W_q     = (const float*)d_in[11];
  const float* phi_w1  = (const float*)d_in[12];
  const float* phi_b1  = (const float*)d_in[13];
  const float* phi_w2  = (const float*)d_in[14];
  const float* phi_b2  = (const float*)d_in[15];
  const float* rho_w1  = (const float*)d_in[16];
  const float* rho_b1  = (const float*)d_in[17];
  const float* rho_w2  = (const float*)d_in[18];
  const float* rho_b2  = (const float*)d_in[19];
  float* out = (float*)d_out;

  char* base = (char*)d_ws;
  size_t off = 0;
  auto take = [&](size_t bytes) -> void* {
    void* p = base + off;
    off += (bytes + 255) & ~(size_t)255;
    return p;
  };
  _Float16* comb_swz = (_Float16*)take((size_t)NTILES * 512 * 2);
  float*    enc_psi  = (float*)take((size_t)TOK * 32 * 4);
  float*    enc_phi  = (float*)take((size_t)TOK * 32 * 4);
  _Float16* agg_swz  = (_Float16*)take((size_t)NTILES * 512 * 2);
  float*    preattn  = (float*)take((size_t)TOK * HEADS * 4);
  _Float16* agg2_swz = (_Float16*)take((size_t)NTILES * 2048 * 2);
  _Float16* wf       = (_Float16*)take((size_t)F_TOTAL * 2);

  const int GEMM_BLOCKS = NTILES / 4;  // 128 blocks * 4 waves

  prep_weights<<<1, 256, 0, stream>>>(psi_w1, psi_w2, phi_w1, phi_w2, arho_w,
                                      W_k, W_q, rho_w1, rho_w2, wf);
  encode_kernel<<<(TOK + 255) / 256, 256, 0, stream>>>(times, values, meas,
                                                       mask, comb_swz);
  mlp_psi_phi<<<GEMM_BLOCKS, 128, 0, stream>>>(comb_swz, mask, wf,
                                               psi_b1, psi_b2, phi_b1, phi_b2,
                                               enc_psi, enc_phi);
  scan_agg<<<BB, 256, 0, stream>>>(enc_psi, mask, agg_swz);
  key_preattn<<<GEMM_BLOCKS, 128, 0, stream>>>(comb_swz, agg_swz, mask, wf,
                                               arho_b, preattn);
  scan_attn<<<BB * HEADS, 256, 0, stream>>>(preattn, enc_phi, mask, agg2_swz);
  rho_mlp<<<GEMM_BLOCKS, 128, 0, stream>>>(agg2_swz, mask, wf, rho_b1, rho_b2,
                                           out);
}
